// DebertaMoEModel_31894427140184
// MI455X (gfx1250) — compile-verified, matrix operand order checked by reading, not run on hardware
//
#include <hip/hip_runtime.h>
#include <stdint.h>

// ---------------- problem constants ----------------
#define B_TOK 4096
#define H_DIM 1024
#define I_DIM 4096
#define TWO_I 8192
#define E_NUM 8
#define NPAIR (B_TOK * 2)

// ---------------- GEMM tiling ----------------
#define BM 128
#define BN 128
#define BK 32
#define LDT 40   // LDS row stride in halves (80 bytes: 16B-aligned, conflict-free)

typedef __attribute__((ext_vector_type(16))) __bf16 v16bf;
typedef __attribute__((ext_vector_type(8)))  float  v8f;
typedef __attribute__((ext_vector_type(2)))  __bf16 v2bf;

union Frag16 { v16bf v; uint4 q[2]; };

// pack two f32 -> packed bf16 dword (native cvt; lowers to v_cvt_pk_bf16_f32 / v_cvt_bf16)
__device__ __forceinline__ unsigned int pack_bf16(float a, float b) {
    v2bf t; t.x = (__bf16)a; t.y = (__bf16)b;
    return __builtin_bit_cast(unsigned int, t);
}

__device__ __forceinline__ v8f v8f_zero() {
    v8f z = {0.f, 0.f, 0.f, 0.f, 0.f, 0.f, 0.f, 0.f};
    return z;
}

__device__ __forceinline__ int imin(int a, int b) { return a < b ? a : b; }

// ============ kernel 1: zero output + routing counters ============
__global__ __launch_bounds__(256) void zero_init_kernel(float4* out4, int n4,
                                                        int* counts, int* fillc) {
    int gid = blockIdx.x * 256 + threadIdx.x;
    if (gid < n4) {
        float4 z; z.x = 0.f; z.y = 0.f; z.z = 0.f; z.w = 0.f;
        out4[gid] = z;
    }
    if (blockIdx.x == 0 && threadIdx.x < E_NUM) {
        counts[threadIdx.x] = 0;
        fillc[threadIdx.x]  = 0;
    }
}

// ============ kernel 2: router (one wave32 per token) ============
__global__ __launch_bounds__(256) void router_kernel(const float* __restrict__ x,
                                                     const float* __restrict__ noise,
                                                     const float* __restrict__ Wr,
                                                     const float* __restrict__ br,
                                                     int* __restrict__ topk_idx,
                                                     float* __restrict__ topk_gate,
                                                     int* __restrict__ counts) {
    const int wave = threadIdx.x >> 5;
    const int lane = threadIdx.x & 31;
    const int b = blockIdx.x * 8 + wave;
    if (b >= B_TOK) return;

    float acc[E_NUM];
#pragma unroll
    for (int e = 0; e < E_NUM; ++e) acc[e] = 0.f;

    for (int i = 0; i < H_DIM / 32; ++i) {
        const int h = i * 32 + lane;
        const float xv = x[(size_t)b * H_DIM + h];
        const float* wr = Wr + (size_t)h * E_NUM;
#pragma unroll
        for (int e = 0; e < E_NUM; ++e) acc[e] += xv * wr[e];
    }
#pragma unroll
    for (int e = 0; e < E_NUM; ++e) {
#pragma unroll
        for (int m = 16; m >= 1; m >>= 1) acc[e] += __shfl_xor(acc[e], m, 32);
    }
    if (lane == 0) {
        float v1 = -3.4e38f, v2 = -3.4e38f;
        int i1 = 0, i2 = 0;
#pragma unroll
        for (int e = 0; e < E_NUM; ++e) {
            float y = acc[e] + br[e] + noise[(size_t)b * E_NUM + e];
            if (y > v1) { v2 = v1; i2 = i1; v1 = y; i1 = e; }
            else if (y > v2) { v2 = y; i2 = e; }
        }
        float w1 = 1.f / (1.f + __expf(v2 - v1));
        float w2 = 1.f - w1;
        topk_idx[b * 2 + 0] = i1;  topk_gate[b * 2 + 0] = w1;
        topk_idx[b * 2 + 1] = i2;  topk_gate[b * 2 + 1] = w2;
        atomicAdd(&counts[i1], 1);
        atomicAdd(&counts[i2], 1);
    }
}

// ============ kernel 3: prefix-sum offsets (8 values) ============
__global__ void offsets_kernel(const int* __restrict__ counts, int* __restrict__ offs) {
    if (threadIdx.x == 0 && blockIdx.x == 0) {
        int s = 0;
        for (int e = 0; e < E_NUM; ++e) { offs[e] = s; s += counts[e]; }
    }
}

// ============ kernel 4: compact (token,gate) pairs per expert ============
__global__ __launch_bounds__(256) void fill_kernel(const int* __restrict__ topk_idx,
                                                   const float* __restrict__ topk_gate,
                                                   const int* __restrict__ offs,
                                                   int* __restrict__ fillc,
                                                   int* __restrict__ pair_token,
                                                   float* __restrict__ pair_gate) {
    const int b = blockIdx.x * 256 + threadIdx.x;
    if (b >= B_TOK) return;
#pragma unroll
    for (int k = 0; k < 2; ++k) {
        const int e = topk_idx[b * 2 + k];
        const int p = atomicAdd(&fillc[e], 1);
        const int s = offs[e] + p;
        pair_token[s] = b;
        pair_gate[s]  = topk_gate[b * 2 + k];
    }
}

// ---------- fragment loader (ISA §7.12.2 layouts) ----------
__device__ __forceinline__ v16bf load_frag(const unsigned short* base, int row, int lane) {
    const int g = (lane >> 4) * 8;
    const unsigned short* p = base + (size_t)row * LDT + g;
    Frag16 u;
    u.q[0] = *(const uint4*)(p);
    u.q[1] = *(const uint4*)(p + 16);
    return u.v;
}

// ============ kernel 5: GEMM1 (x @ W_in) + SwiGLU -> a_buf (bf16) ============
// Double-buffered LDS; fp32 -> packed bf16 on the staging path.
__global__ __launch_bounds__(256) void gemm1_swiglu_kernel(
        const float* __restrict__ x,
        const float* __restrict__ W_in,
        const float* __restrict__ b_in,
        const int* __restrict__ offs,
        const int* __restrict__ counts,
        const int* __restrict__ pair_token,
        unsigned short* __restrict__ a_buf) {
    const int e  = blockIdx.y >> 5;
    const int mt = blockIdx.y & 31;
    const int cnt = counts[e];
    if (mt * BM >= cnt) return;
    const int nb = blockIdx.x * BN;                 // column in [0, I)
    const int slot_base = offs[e] + mt * BM;
    const int cnt_loc = cnt - mt * BM;              // valid rows in this tile

    __shared__ unsigned short sA [2][BM * LDT];
    __shared__ unsigned short sB1[2][BN * LDT];     // transposed: [n][k]
    __shared__ unsigned short sB2[2][BN * LDT];

    const int tid  = threadIdx.x;
    const int lane = tid & 31;
    const int wave = tid >> 5;
    const int wm = (wave >> 2) * 64;                // 2 wave-rows of 64
    const int wn = (wave & 3) * 32;                 // 4 wave-cols of 32

    // tokens for the 4 A-rows this thread stages (clamped; junk rows never stored)
    int myTok[4];
#pragma unroll
    for (int i = 0; i < 4; ++i) {
        const int r = (tid >> 3) + 32 * i;
        myTok[i] = pair_token[imin(slot_base + r, NPAIR - 1)];
    }

    const float* Wb = W_in + (size_t)e * H_DIM * TWO_I;

    v8f acc1[4][2], acc2[4][2];
#pragma unroll
    for (int mi = 0; mi < 4; ++mi)
#pragma unroll
        for (int ni = 0; ni < 2; ++ni) { acc1[mi][ni] = v8f_zero(); acc2[mi][ni] = v8f_zero(); }

    float4 ra[4];            // staged A
    float4 rb1[2][2], rb2[2][2];  // staged B1/B2 (two K-rows per item)

    const int a_k0 = (tid & 7) * 4;        // A: k offset within BK
    const int b_n0 = (tid & 31) * 4;       // B: n offset within BN

#define G1_LOAD(KK)                                                                   \
    {                                                                                 \
        _Pragma("unroll")                                                             \
        for (int i = 0; i < 4; ++i)                                                   \
            ra[i] = *(const float4*)(x + (size_t)myTok[i] * H_DIM + (KK) + a_k0);     \
        _Pragma("unroll")                                                             \
        for (int i = 0; i < 2; ++i) {                                                 \
            const int kp = (tid >> 5) + 8 * i;                                        \
            const float* s0 = Wb + (size_t)((KK) + 2 * kp) * TWO_I + nb + b_n0;       \
            rb1[i][0] = *(const float4*)(s0);                                         \
            rb1[i][1] = *(const float4*)(s0 + TWO_I);                                 \
            rb2[i][0] = *(const float4*)(s0 + I_DIM);                                 \
            rb2[i][1] = *(const float4*)(s0 + I_DIM + TWO_I);                         \
        }                                                                             \
    }

#define G1_STORE(BUF)                                                                 \
    {                                                                                 \
        _Pragma("unroll")                                                             \
        for (int i = 0; i < 4; ++i) {                                                 \
            const int r = (tid >> 3) + 32 * i;                                        \
            uint2 p;                                                                  \
            p.x = pack_bf16(ra[i].x, ra[i].y);                                        \
            p.y = pack_bf16(ra[i].z, ra[i].w);                                        \
            *(uint2*)&sA[BUF][(size_t)r * LDT + a_k0] = p;                            \
        }                                                                             \
        _Pragma("unroll")                                                             \
        for (int i = 0; i < 2; ++i) {                                                 \
            const int kp2 = ((tid >> 5) + 8 * i) * 2;                                 \
            *(unsigned int*)&sB1[BUF][(size_t)(b_n0 + 0) * LDT + kp2] = pack_bf16(rb1[i][0].x, rb1[i][1].x); \
            *(unsigned int*)&sB1[BUF][(size_t)(b_n0 + 1) * LDT + kp2] = pack_bf16(rb1[i][0].y, rb1[i][1].y); \
            *(unsigned int*)&sB1[BUF][(size_t)(b_n0 + 2) * LDT + kp2] = pack_bf16(rb1[i][0].z, rb1[i][1].z); \
            *(unsigned int*)&sB1[BUF][(size_t)(b_n0 + 3) * LDT + kp2] = pack_bf16(rb1[i][0].w, rb1[i][1].w); \
            *(unsigned int*)&sB2[BUF][(size_t)(b_n0 + 0) * LDT + kp2] = pack_bf16(rb2[i][0].x, rb2[i][1].x); \
            *(unsigned int*)&sB2[BUF][(size_t)(b_n0 + 1) * LDT + kp2] = pack_bf16(rb2[i][0].y, rb2[i][1].y); \
            *(unsigned int*)&sB2[BUF][(size_t)(b_n0 + 2) * LDT + kp2] = pack_bf16(rb2[i][0].z, rb2[i][1].z); \
            *(unsigned int*)&sB2[BUF][(size_t)(b_n0 + 3) * LDT + kp2] = pack_bf16(rb2[i][0].w, rb2[i][1].w); \
        }                                                                             \
    }

    // prologue: stage K-step 0 into buffer 0
    G1_LOAD(0)
    G1_STORE(0)
    __syncthreads();

    int buf = 0;
    for (int kk = 0; kk < H_DIM; kk += BK, buf ^= 1) {
        const bool has_next = (kk + BK) < H_DIM;
        if (has_next) G1_LOAD(kk + BK)          // global loads in flight over WMMAs

        // ---- WMMA on current buffer ----
        v16bf aF[4];
#pragma unroll
        for (int mi = 0; mi < 4; ++mi)
            aF[mi] = load_frag(sA[buf], wm + mi * 16 + (lane & 15), lane);
#pragma unroll
        for (int ni = 0; ni < 2; ++ni) {
            const int N = wn + ni * 16 + (lane & 15);
            v16bf b1 = load_frag(sB1[buf], N, lane);
            v16bf b2 = load_frag(sB2[buf], N, lane);
#pragma unroll
            for (int mi = 0; mi < 4; ++mi) {
                acc1[mi][ni] = __builtin_amdgcn_wmma_f32_16x16x32_bf16(
                    false, aF[mi], false, b1, (short)0, acc1[mi][ni], false, false);
                acc2[mi][ni] = __builtin_amdgcn_wmma_f32_16x16x32_bf16(
                    false, aF[mi], false, b2, (short)0, acc2[mi][ni], false, false);
            }
        }
        if (has_next) G1_STORE(buf ^ 1)
        __syncthreads();
    }
#undef G1_LOAD
#undef G1_STORE

    // ---- epilogue: h = acc + b_in ; a = h1 * sigmoid(h2) -> bf16 a_buf ----
    const float* bi = b_in + (size_t)e * TWO_I;
    const int g8 = ((lane >> 4) << 3);
#pragma unroll
    for (int mi = 0; mi < 4; ++mi) {
#pragma unroll
        for (int ni = 0; ni < 2; ++ni) {
            const int gcol = nb + wn + ni * 16 + (lane & 15);
            const float bi1 = bi[gcol];
            const float bi2 = bi[I_DIM + gcol];
#pragma unroll
            for (int v = 0; v < 8; ++v) {
                const int row = wm + mi * 16 + g8 + v;   // M = v + 8*g
                if (row < cnt_loc) {
                    const float h1 = acc1[mi][ni][v] + bi1;
                    const float h2 = acc2[mi][ni][v] + bi2;
                    const float a  = h1 * (1.f / (1.f + __expf(-h2)));
                    a_buf[(size_t)(slot_base + row) * I_DIM + gcol] =
                        __builtin_bit_cast(unsigned short, (__bf16)a);
                }
            }
        }
    }
}

// ============ kernel 6: GEMM2 (a @ W_out), gate-scale, scatter-add ============
__global__ __launch_bounds__(256) void gemm2_scatter_kernel(
        const unsigned short* __restrict__ a_buf,
        const float* __restrict__ W_out,
        const float* __restrict__ b_out,
        const int* __restrict__ offs,
        const int* __restrict__ counts,
        const int* __restrict__ pair_token,
        const float* __restrict__ pair_gate,
        float* __restrict__ out) {
    const int e  = blockIdx.y >> 5;
    const int mt = blockIdx.y & 31;
    const int cnt = counts[e];
    if (mt * BM >= cnt) return;
    const int nb = blockIdx.x * BN;                 // column in [0, H)
    const int slot_base = offs[e] + mt * BM;
    const int cnt_loc = cnt - mt * BM;

    __shared__ unsigned short sA[2][BM * LDT];
    __shared__ unsigned short sB[2][BN * LDT];      // transposed: [n][k]
    __shared__ int   sTok [BM];
    __shared__ float sGate[BM];

    const int tid  = threadIdx.x;
    const int lane = tid & 31;
    const int wave = tid >> 5;
    const int wm = (wave >> 2) * 64;
    const int wn = (wave & 3) * 32;

    if (tid < BM) {
        const bool valid = tid < cnt_loc;
        const int s = imin(slot_base + tid, NPAIR - 1);
        sTok [tid] = valid ? pair_token[s] : -1;
        sGate[tid] = valid ? pair_gate [s] : 0.f;
    }

    const float* Wb = W_out + (size_t)e * I_DIM * H_DIM;

    v8f acc[4][2];
#pragma unroll
    for (int mi = 0; mi < 4; ++mi)
#pragma unroll
        for (int ni = 0; ni < 2; ++ni) acc[mi][ni] = v8f_zero();

    uint4  rA[2];
    float4 rb[2][2];
    const int a_r = tid >> 2;
    const int a_q = tid & 3;
    const int b_n0 = (tid & 31) * 4;

#define G2_LOAD(KK)                                                                   \
    {                                                                                 \
        _Pragma("unroll")                                                             \
        for (int i = 0; i < 2; ++i) {                                                 \
            const int r = a_r + 64 * i;  /* a_buf over-allocated: always in-bounds */ \
            rA[i] = *(const uint4*)(a_buf + (size_t)(slot_base + r) * I_DIM + (KK) + a_q * 8); \
        }                                                                             \
        _Pragma("unroll")                                                             \
        for (int i = 0; i < 2; ++i) {                                                 \
            const int kp = (tid >> 5) + 8 * i;                                        \
            const float* s0 = Wb + (size_t)((KK) + 2 * kp) * H_DIM + nb + b_n0;       \
            rb[i][0] = *(const float4*)(s0);                                          \
            rb[i][1] = *(const float4*)(s0 + H_DIM);                                  \
        }                                                                             \
    }

#define G2_STORE(BUF)                                                                 \
    {                                                                                 \
        _Pragma("unroll")                                                             \
        for (int i = 0; i < 2; ++i)                                                   \
            *(uint4*)&sA[BUF][(size_t)(a_r + 64 * i) * LDT + a_q * 8] = rA[i];        \
        _Pragma("unroll")                                                             \
        for (int i = 0; i < 2; ++i) {                                                 \
            const int kp2 = ((tid >> 5) + 8 * i) * 2;                                 \
            *(unsigned int*)&sB[BUF][(size_t)(b_n0 + 0) * LDT + kp2] = pack_bf16(rb[i][0].x, rb[i][1].x); \
            *(unsigned int*)&sB[BUF][(size_t)(b_n0 + 1) * LDT + kp2] = pack_bf16(rb[i][0].y, rb[i][1].y); \
            *(unsigned int*)&sB[BUF][(size_t)(b_n0 + 2) * LDT + kp2] = pack_bf16(rb[i][0].z, rb[i][1].z); \
            *(unsigned int*)&sB[BUF][(size_t)(b_n0 + 3) * LDT + kp2] = pack_bf16(rb[i][0].w, rb[i][1].w); \
        }                                                                             \
    }

    G2_LOAD(0)
    G2_STORE(0)
    __syncthreads();

    int buf = 0;
    for (int kk = 0; kk < I_DIM; kk += BK, buf ^= 1) {
        const bool has_next = (kk + BK) < I_DIM;
        if (has_next) G2_LOAD(kk + BK)

        v16bf aF[4];
#pragma unroll
        for (int mi = 0; mi < 4; ++mi)
            aF[mi] = load_frag(sA[buf], wm + mi * 16 + (lane & 15), lane);
#pragma unroll
        for (int ni = 0; ni < 2; ++ni) {
            v16bf bF = load_frag(sB[buf], wn + ni * 16 + (lane & 15), lane);
#pragma unroll
            for (int mi = 0; mi < 4; ++mi)
                acc[mi][ni] = __builtin_amdgcn_wmma_f32_16x16x32_bf16(
                    false, aF[mi], false, bF, (short)0, acc[mi][ni], false, false);
        }
        if (has_next) G2_STORE(buf ^ 1)
        __syncthreads();
    }
#undef G2_LOAD
#undef G2_STORE

    // ---- epilogue: out[token, col] += gate * (acc + b_out[e, col]) ----
    const float* bo = b_out + (size_t)e * H_DIM;
    const int g8 = ((lane >> 4) << 3);
#pragma unroll
    for (int mi = 0; mi < 4; ++mi) {
#pragma unroll
        for (int ni = 0; ni < 2; ++ni) {
            const int gcol = nb + wn + ni * 16 + (lane & 15);
            const float bv = bo[gcol];
#pragma unroll
            for (int v = 0; v < 8; ++v) {
                const int row = wm + mi * 16 + g8 + v;
                const int tok = sTok[row];
                if (tok >= 0) {
                    const float val = sGate[row] * (acc[mi][ni][v] + bv);
                    unsafeAtomicAdd(&out[(size_t)tok * H_DIM + gcol], val);
                }
            }
        }
    }
}

// ============================ launcher ============================
extern "C" void kernel_launch(void* const* d_in, const int* in_sizes, int n_in,
                              void* d_out, int out_size, void* d_ws, size_t ws_size,
                              hipStream_t stream) {
    const float* x     = (const float*)d_in[0];
    const float* noise = (const float*)d_in[1];
    const float* Wr    = (const float*)d_in[2];
    const float* br    = (const float*)d_in[3];
    const float* W_in  = (const float*)d_in[4];
    const float* b_in  = (const float*)d_in[5];
    const float* W_out = (const float*)d_in[6];
    const float* b_out = (const float*)d_in[7];
    float* out = (float*)d_out;

    // workspace layout
    unsigned char* w = (unsigned char*)d_ws;
    int*   counts     = (int*)(w + 0);
    int*   fillc      = (int*)(w + 32);
    int*   offs       = (int*)(w + 64);
    int*   topk_idx   = (int*)(w + 128);
    float* topk_gate  = (float*)(w + 128 + NPAIR * 4);
    int*   pair_token = (int*)(w + 128 + NPAIR * 8);
    float* pair_gate  = (float*)(w + 128 + NPAIR * 12);
    size_t abuf_off   = ((size_t)(128 + NPAIR * 16) + 255) & ~(size_t)255;
    unsigned short* a_buf = (unsigned short*)(w + abuf_off);   // (NPAIR+128) x I bf16

    const int n4 = (B_TOK * H_DIM) / 4;
    zero_init_kernel<<<n4 / 256, 256, 0, stream>>>((float4*)out, n4, counts, fillc);
    router_kernel<<<B_TOK / 8, 256, 0, stream>>>(x, noise, Wr, br, topk_idx, topk_gate, counts);
    offsets_kernel<<<1, 32, 0, stream>>>(counts, offs);
    fill_kernel<<<B_TOK / 256, 256, 0, stream>>>(topk_idx, topk_gate, offs, fillc,
                                                 pair_token, pair_gate);
    dim3 g1(I_DIM / BN, E_NUM * 32);
    gemm1_swiglu_kernel<<<g1, 256, 0, stream>>>(x, W_in, b_in, offs, counts, pair_token, a_buf);
    dim3 g2(H_DIM / BN, E_NUM * 32);
    gemm2_scatter_kernel<<<g2, 256, 0, stream>>>(a_buf, W_out, b_out, offs, counts,
                                                 pair_token, pair_gate, out);
}
